// ReLUKAN_55860344652231
// MI455X (gfx1250) — matrix operand order repaired and checked by reading, not compiled, not run on hardware
//
#include <hip/hip_runtime.h>
#include <hip/hip_bf16.h>

#define BATCH 4096
#define IN_DIM 256
#define GRIDN 64
#define HID 128
#define DEPTH 3

typedef __bf16 v16bf __attribute__((ext_vector_type(16)));
typedef float  v8f   __attribute__((ext_vector_type(8)));
typedef float  v2f   __attribute__((ext_vector_type(2)));
typedef int    v4i   __attribute__((ext_vector_type(4)));

typedef __attribute__((address_space(1))) v4i* gv4i_t;  // global int4*
typedef __attribute__((address_space(3))) v4i* lv4i_t;  // LDS int4*

// ---- CDNA5 async global->LDS copy (ASYNCcnt path), with fallbacks ---------
#if __has_builtin(__builtin_amdgcn_global_load_async_to_lds_b128)
#define ASYNC_CP16(dst, src)                                                  \
  __builtin_amdgcn_global_load_async_to_lds_b128((gv4i_t)(src),               \
                                                 (lv4i_t)(dst), 0, 0)
#else
#define ASYNC_CP16(dst, src)                                                  \
  asm volatile("global_load_async_to_lds_b128 %0, %1, off" ::                 \
                   "v"(dst), "v"(src) : "memory")
#endif

#if __has_builtin(__builtin_amdgcn_s_wait_asynccnt)
#define WAIT_ASYNC() __builtin_amdgcn_s_wait_asynccnt(0)
#else
#define WAIT_ASYNC() asm volatile("s_wait_asynccnt 0" ::: "memory")
#endif

// ---------------------------------------------------------------------------
// per-feature min / scale over the batch  (mn[i], rs[i] = 1/(max-min+1e-6))
// ---------------------------------------------------------------------------
__global__ __launch_bounds__(256) void minmax_kernel(
    const float* __restrict__ x, float* __restrict__ mn, float* __restrict__ rs) {
  const int i = blockIdx.x;
  const int tid = threadIdx.x;
  float lo = 3.4e38f, hi = -3.4e38f;
  for (int b = tid; b < BATCH; b += 256) {
    float v = x[(size_t)b * IN_DIM + i];
    lo = fminf(lo, v);
    hi = fmaxf(hi, v);
  }
  __shared__ float rlo[256], rhi[256];
  rlo[tid] = lo; rhi[tid] = hi;
  __syncthreads();
  for (int off = 128; off > 0; off >>= 1) {
    if (tid < off) {
      rlo[tid] = fminf(rlo[tid], rlo[tid + off]);
      rhi[tid] = fmaxf(rhi[tid], rhi[tid + off]);
    }
    __syncthreads();
  }
  if (tid == 0) {
    mn[i] = rlo[0];
    rs[i] = 1.0f / (rhi[0] - rlo[0] + 1e-6f);
  }
}

__global__ __launch_bounds__(256) void zero_kernel(float* __restrict__ p, int n) {
  int idx = blockIdx.x * 256 + threadIdx.x;
  if (idx < n) p[idx] = 0.0f;
}

// ---------------------------------------------------------------------------
// Prepass: fs[pair][g][n] f32  ->  fst[pair][n*72+g] bf16 (transposed+padded)
// Row byte stride 144 (=9*16) keeps every B-fragment ds_load_b128 aligned.
// ---------------------------------------------------------------------------
#define FS_PAD 72
#define PAIR_ELEMS (HID * FS_PAD)
__global__ __launch_bounds__(256) void convert_fs_kernel(
    const float* __restrict__ fs, __bf16* __restrict__ fst) {
  const int pair = blockIdx.x;
  const float* __restrict__ src = fs + (size_t)pair * (GRIDN * HID);
  __bf16* __restrict__ dst = fst + (size_t)pair * PAIR_ELEMS;
  for (int idx = threadIdx.x; idx < GRIDN * HID; idx += 256) {
    int g = idx >> 7, n = idx & 127;
    dst[n * FS_PAD + g] = (__bf16)src[idx];
  }
}

// ---------------------------------------------------------------------------
// KAN basis einsum:  h[b,n] += sum_{d,i,g} relu(z[b,i]-grid[d,g]) * fs[d,i,g,n]
// bf16 WMMA 16x16x32, f32 accumulate. Double-buffered async LDS staging;
// x-scalar prefetch one tile ahead; B-fragments batch-loaded per k-step.
// ---------------------------------------------------------------------------
__global__ __launch_bounds__(256) void kan_kernel(
    const float* __restrict__ x, const __bf16* __restrict__ fst,
    const float* __restrict__ grids, const float* __restrict__ mn,
    const float* __restrict__ rs, float* __restrict__ h) {
  __shared__ __align__(16) __bf16 fsb[2][PAIR_ELEMS];  // 2 x 18 KB
  __shared__ float glds[DEPTH * GRIDN];

  const int tid = threadIdx.x;
  const int lane = tid & 31;
  const int wv = tid >> 5;
  const int hw = lane >> 4;
  const int ln = lane & 15;
  const int mBase = blockIdx.x * 128;
  const int row = mBase + wv * 16 + ln;

  for (int t = tid; t < DEPTH * GRIDN; t += 256) glds[t] = grids[t];

  v8f acc[8];
#pragma unroll
  for (int t = 0; t < 8; ++t)
#pragma unroll
    for (int r = 0; r < 8; ++r) acc[t][r] = 0.0f;

  const int pairBeg = blockIdx.y * 96;  // 768 (d,i) pairs / 8 splits
  const int NCHUNK = PAIR_ELEMS * 2 / 16;  // 1152 x 16B per tile

  // prologue: DMA of first tile + x scalar for first pair
  {
    const __bf16* src = fst + (size_t)pairBeg * PAIR_ELEMS;
    for (int c = tid; c < NCHUNK; c += 256)
      ASYNC_CP16(&fsb[0][c * 8], src + c * 8);
  }
  int i0 = pairBeg & 255;
  float zv = (x[(size_t)row * IN_DIM + i0] - mn[i0]) * rs[i0];

  for (int p = 0; p < 96; ++p) {
    const int cur = p & 1;
    const int pair = pairBeg + p;
    const int d = pair >> 8;

    WAIT_ASYNC();        // own async copies for tile p complete
    __syncthreads();     // ... and everyone else's; buf[cur^1] free to reuse

    if (p + 1 < 96) {    // issue DMA for next tile, overlapped with compute
      const __bf16* src = fst + (size_t)(pair + 1) * PAIR_ELEMS;
      for (int c = tid; c < NCHUNK; c += 256)
        ASYNC_CP16(&fsb[cur ^ 1][c * 8], src + c * 8);
    }

    // prefetch next tile's z scalar (hides VMEM latency behind the WMMAs)
    float zv_next = 0.0f;
    if (p + 1 < 96) {
      int inext = (pair + 1) & 255;
      zv_next = (x[(size_t)row * IN_DIM + inext] - mn[inext]) * rs[inext];
    }

    // build both A fragments up front (contiguous k = e + 16*hw, same order as B)
    v16bf afr[2];
#pragma unroll
    for (int kk2 = 0; kk2 < 2; ++kk2)
#pragma unroll
      for (int e = 0; e < 16; ++e) {
        int k = e + (hw ? 16 : 0);
        float av = zv - glds[d * GRIDN + kk2 * 32 + k];
        afr[kk2][e] = (__bf16)(av > 0.0f ? av : 0.0f);
      }

#pragma unroll
    for (int kk2 = 0; kk2 < 2; ++kk2) {
      // batch-load all 8 B fragments, then fire 8 WMMAs back-to-back
      v16bf bfr[8];
#pragma unroll
      for (int t = 0; t < 8; ++t) {
        const __bf16* bp =
            &fsb[cur][(t * 16 + ln) * FS_PAD + kk2 * 32 + hw * 16];
        __builtin_memcpy(&bfr[t], bp, 32);
      }
#pragma unroll
      for (int t = 0; t < 8; ++t)
        acc[t] = __builtin_amdgcn_wmma_f32_16x16x32_bf16(
            false, afr[kk2], false, bfr[t], (short)0, acc[t], false, false);
    }
    zv = zv_next;
  }

#pragma unroll
  for (int t = 0; t < 8; ++t)
#pragma unroll
    for (int r = 0; r < 8; ++r) {
      int m = mBase + wv * 16 + r + hw * 8;
      int n = t * 16 + ln;
      unsafeAtomicAdd(&h[(size_t)m * HID + n], acc[t][r]);
    }
}

// ---------------------------------------------------------------------------
// MLP layer: out = [relu]( BN(in) @ W + bias ), BN folded to sc/sh (nullable).
// fp32 WMMA 16x16x4. W tile staged via async DMA; A tile staged via VALU
// (BN scale/shift folded during the copy).
// ---------------------------------------------------------------------------
#define A_PAD 68
__global__ __launch_bounds__(256) void mlp_gemm(
    const float* __restrict__ in, const float* __restrict__ sc,
    const float* __restrict__ sh, const float* __restrict__ W,
    const float* __restrict__ bias, float* __restrict__ out,
    int K, int N, int relu) {
  __shared__ __align__(16) float a_s[128 * A_PAD];
  __shared__ __align__(16) float w_s[64 * A_PAD];
  const int tid = threadIdx.x;
  const int lane = tid & 31;
  const int wv = tid >> 5;
  const int hw = lane >> 4;
  const int ln = lane & 15;
  const int mBase = blockIdx.x * 128;
  const int nBase = blockIdx.y * 64;

  v8f acc[4];
#pragma unroll
  for (int t = 0; t < 4; ++t)
#pragma unroll
    for (int r = 0; r < 8; ++r) acc[t][r] = 0.0f;

  for (int kc = 0; kc < K; kc += 64) {
    __syncthreads();
    // async DMA: 64x64 f32 W tile, 16B chunks (row byte stride 272 = 17*16)
    for (int c = tid; c < 64 * 16; c += 256) {
      int kk = c >> 4, q = c & 15;
      ASYNC_CP16(&w_s[kk * A_PAD + q * 4],
                 W + (size_t)(kc + kk) * N + nBase + q * 4);
    }
    // VALU staging of A with folded BatchNorm of the previous layer
    for (int idx = tid; idx < 128 * 64; idx += 256) {
      int m = idx >> 6, kk = idx & 63;
      int k = kc + kk;
      float v = in[(size_t)(mBase + m) * K + k];
      if (sc) v = v * sc[k] + sh[k];
      a_s[m * A_PAD + kk] = v;
    }
    WAIT_ASYNC();
    __syncthreads();

    for (int ks = 0; ks < 64; ks += 4) {
      v2f a;
      a[0] = a_s[(wv * 16 + ln) * A_PAD + ks + 2 * hw + 0];
      a[1] = a_s[(wv * 16 + ln) * A_PAD + ks + 2 * hw + 1];
      v2f b[4];
#pragma unroll
      for (int t = 0; t < 4; ++t) {
        b[t][0] = w_s[(ks + 2 * hw + 0) * A_PAD + t * 16 + ln];
        b[t][1] = w_s[(ks + 2 * hw + 1) * A_PAD + t * 16 + ln];
      }
#pragma unroll
      for (int t = 0; t < 4; ++t)
        acc[t] = __builtin_amdgcn_wmma_f32_16x16x4_f32(
            false, a, false, b[t], (short)0, acc[t], false, false);
    }
  }

#pragma unroll
  for (int t = 0; t < 4; ++t)
#pragma unroll
    for (int r = 0; r < 8; ++r) {
      int n = nBase + t * 16 + ln;
      int m = mBase + wv * 16 + r + hw * 8;
      float v = acc[t][r] + bias[n];
      if (relu) v = fmaxf(v, 0.0f);
      out[(size_t)m * N + n] = v;
    }
}

// ---------------------------------------------------------------------------
// Column BN statistics -> folded scale/shift: sc=g/sqrt(var+eps), sh=be-mean*sc
// ---------------------------------------------------------------------------
__global__ __launch_bounds__(256) void colstats_kernel(
    const float* __restrict__ a, const float* __restrict__ g,
    const float* __restrict__ be, float* __restrict__ sc,
    float* __restrict__ sh, int N) {
  const int n = blockIdx.x;
  const int tid = threadIdx.x;
  float s = 0.0f, s2 = 0.0f;
  for (int b = tid; b < BATCH; b += 256) {
    float v = a[(size_t)b * N + n];
    s += v;
    s2 += v * v;
  }
  __shared__ float r0[256], r1[256];
  r0[tid] = s; r1[tid] = s2;
  __syncthreads();
  for (int off = 128; off > 0; off >>= 1) {
    if (tid < off) { r0[tid] += r0[tid + off]; r1[tid] += r1[tid + off]; }
    __syncthreads();
  }
  if (tid == 0) {
    float mean = r0[0] * (1.0f / BATCH);
    float var = r1[0] * (1.0f / BATCH) - mean * mean;
    float s_ = g[n] * rsqrtf(var + 1e-5f);
    sc[n] = s_;
    sh[n] = be[n] - mean * s_;
  }
}

// ---------------------------------------------------------------------------
// Final head: out[b] = b4 + sum_k (a3[b,k]*sc3[k]+sh3[k]) * w4[k]
// ---------------------------------------------------------------------------
__global__ __launch_bounds__(256) void final_kernel(
    const float* __restrict__ a3, const float* __restrict__ sc,
    const float* __restrict__ sh, const float* __restrict__ w4,
    const float* __restrict__ b4, float* __restrict__ out) {
  int b = blockIdx.x * 256 + threadIdx.x;
  float s = b4[0];
#pragma unroll
  for (int k = 0; k < 64; ++k)
    s += (a3[(size_t)b * 64 + k] * sc[k] + sh[k]) * w4[k];
  out[b] = s;
}

// ---------------------------------------------------------------------------
extern "C" void kernel_launch(void* const* d_in, const int* in_sizes, int n_in,
                              void* d_out, int out_size, void* d_ws, size_t ws_size,
                              hipStream_t stream) {
  const float* x     = (const float*)d_in[0];
  const float* fs    = (const float*)d_in[1];
  const float* grids = (const float*)d_in[2];
  const float* w1 = (const float*)d_in[3];
  const float* b1 = (const float*)d_in[4];
  const float* g1 = (const float*)d_in[5];
  const float* be1 = (const float*)d_in[6];
  const float* w2 = (const float*)d_in[7];
  const float* b2 = (const float*)d_in[8];
  const float* g2 = (const float*)d_in[9];
  const float* be2 = (const float*)d_in[10];
  const float* w3 = (const float*)d_in[11];
  const float* b3 = (const float*)d_in[12];
  const float* g3 = (const float*)d_in[13];
  const float* be3 = (const float*)d_in[14];
  const float* w4 = (const float*)d_in[15];
  const float* b4 = (const float*)d_in[16];
  float* out = (float*)d_out;

  float* w = (float*)d_ws;
  float* mn = w;             // 256
  float* rs = mn + 256;      // 256
  float* h  = rs + 256;      // 4096*128
  float* a1 = h + BATCH * HID;        // 4096*256
  float* a2 = a1 + BATCH * (HID * 2); // 4096*128
  float* a3 = a2 + BATCH * HID;       // 4096*64
  float* sc1 = a3 + BATCH * (HID / 2);
  float* sh1 = sc1 + 256;
  float* sc2 = sh1 + 256;
  float* sh2 = sc2 + 128;
  float* sc3 = sh2 + 128;
  float* sh3 = sc3 + 64;
  __bf16* fst = (__bf16*)(sh3 + 64);  // 768 * 128*72 bf16 (16B-aligned offset)

  convert_fs_kernel<<<DEPTH * IN_DIM, 256, 0, stream>>>(fs, fst);
  minmax_kernel<<<IN_DIM, 256, 0, stream>>>(x, mn, rs);
  zero_kernel<<<(BATCH * HID + 255) / 256, 256, 0, stream>>>(h, BATCH * HID);
  kan_kernel<<<dim3(BATCH / 128, 8), 256, 0, stream>>>(x, fst, grids, mn, rs, h);

  mlp_gemm<<<dim3(BATCH / 128, 4), 256, 0, stream>>>(h, nullptr, nullptr, w1, b1,
                                                     a1, HID, HID * 2, 1);
  colstats_kernel<<<HID * 2, 256, 0, stream>>>(a1, g1, be1, sc1, sh1, HID * 2);

  mlp_gemm<<<dim3(BATCH / 128, 2), 256, 0, stream>>>(a1, sc1, sh1, w2, b2,
                                                     a2, HID * 2, HID, 1);
  colstats_kernel<<<HID, 256, 0, stream>>>(a2, g2, be2, sc2, sh2, HID);

  mlp_gemm<<<dim3(BATCH / 128, 1), 256, 0, stream>>>(a2, sc2, sh2, w3, b3,
                                                     a3, HID, HID / 2, 1);
  colstats_kernel<<<HID / 2, 256, 0, stream>>>(a3, g3, be3, sc3, sh3, HID / 2);

  final_kernel<<<BATCH / 256, 256, 0, stream>>>(a3, sc3, sh3, w4, b4, out);
}
